// LayerEncoderBlock_27968827031952
// MI455X (gfx1250) — compile-verified, matrix operand order checked by reading, not run on hardware
//
#include <hip/hip_runtime.h>

// ---------------------------------------------------------------------------
// CDNA5 (gfx1250) hierarchical window-attention encoder block.
// All GEMM-shaped math runs through v_wmma_f32_16x16x32_bf16 (wave32 WMMA).
// ---------------------------------------------------------------------------

typedef __attribute__((ext_vector_type(16))) __bf16 v16bf;
typedef __attribute__((ext_vector_type(8)))  float  v8f;

#define HEADS 8
#define CDIM  256
#define DH    32

// ----- helpers -------------------------------------------------------------

__device__ __forceinline__ unsigned short f2bf(float x) {
  unsigned u = __float_as_uint(x);
  return (unsigned short)((u + 0x7FFFu + ((u >> 16) & 1u)) >> 16);  // RNE
}
__device__ __forceinline__ unsigned pack2(float a, float b) {
  return (unsigned)f2bf(a) | ((unsigned)f2bf(b) << 16);
}

union FragBF { v16bf v; unsigned u[8]; };
union FragF  { v8f   v; float    f[8]; };

// Load a 16x32 bf16 fragment (A layout; B fragments are loaded from an LDS
// tile stored transposed so the same loader applies).  Per ISA 7.12.2:
// lane l: m = l&15, h = l>>4; dword g<4 holds K = h*8+2g(+1); g>=4 holds
// K = 16 + h*8 + 2(g-4)(+1).  rowStrideU / kOffU are in dwords.
__device__ __forceinline__ v16bf ld_frag(const unsigned* lds, int rowBase,
                                         int rowStrideU, int kOffU, int lane) {
  FragBF fr;
  int m = lane & 15, h = lane >> 4;
  const unsigned* rp = lds + (size_t)(rowBase + m) * rowStrideU + kOffU;
#pragma unroll
  for (int g = 0; g < 4; ++g) {
    fr.u[g]     = rp[h * 4 + g];
    fr.u[4 + g] = rp[8 + h * 4 + g];
  }
  return fr.v;
}

__device__ __forceinline__ v8f wmma_bf(v16bf a, v16bf b, v8f c) {
  return __builtin_amdgcn_wmma_f32_16x16x32_bf16(
      /*neg_a=*/false, a, /*neg_b=*/false, b,
      /*c_mod=*/(short)0, c, /*reuse_a=*/false, /*reuse_b=*/false);
}

// ----- weight conversion ---------------------------------------------------

__global__ __launch_bounds__(256) void f32_to_bf16_kernel(
    const float* __restrict__ in, unsigned short* __restrict__ out, int n) {
  int i = blockIdx.x * 256 + threadIdx.x;
  if (i < n) out[i] = f2bf(in[i]);
}

// ----- layout transposes ---------------------------------------------------

__global__ __launch_bounds__(256) void nchw_to_nhwc_kernel(
    const float* __restrict__ in, float* __restrict__ out,
    int N, int C, int H, int W) {
  size_t idx = (size_t)blockIdx.x * 256 + threadIdx.x;  // NHWC linear (c fastest)
  size_t total = (size_t)N * C * H * W;
  if (idx >= total) return;
  int c = idx % C; size_t t = idx / C;
  int x = t % W;   t /= W;
  int y = t % H;   int n = t / H;
  out[idx] = in[(((size_t)n * C + c) * H + y) * W + x];
}

__global__ __launch_bounds__(256) void nhwc_to_nchw_kernel(
    const float* __restrict__ in, float* __restrict__ out,
    int N, int C, int H, int W) {
  size_t idx = (size_t)blockIdx.x * 256 + threadIdx.x;  // NCHW linear (x fastest)
  size_t total = (size_t)N * C * H * W;
  if (idx >= total) return;
  int x = idx % W; size_t t = idx / W;
  int y = t % H;   t /= H;
  int c = t % C;   int n = t / C;
  out[idx] = in[(((size_t)n * H + y) * W + x) * C + c];
}

// ----- avgpool 4x4 ---------------------------------------------------------

__global__ __launch_bounds__(256) void avgpool4_kernel(
    const float* __restrict__ in, float* __restrict__ out,
    int N, int H, int W) {                       // in: NHWC, out: N,H/4,W/4,C
  int Ho = H / 4, Wo = W / 4;
  size_t idx = (size_t)blockIdx.x * 256 + threadIdx.x;
  size_t total = (size_t)N * Ho * Wo * CDIM;
  if (idx >= total) return;
  int c = idx % CDIM; size_t t = idx / CDIM;
  int x1 = t % Wo; t /= Wo;
  int y1 = t % Ho; int n = t / Ho;
  float s = 0.f;
#pragma unroll
  for (int dy = 0; dy < 4; ++dy)
#pragma unroll
    for (int dx = 0; dx < 4; ++dx)
      s += in[(((size_t)n * H + y1 * 4 + dy) * W + x1 * 4 + dx) * CDIM + c];
  out[idx] = s * (1.0f / 16.0f);
}

// ----- layernorm (wave per token, C = 256) ---------------------------------

__global__ __launch_bounds__(256) void layernorm_kernel(
    const float* __restrict__ in, float* __restrict__ out,
    const float* __restrict__ g, const float* __restrict__ b, int L) {
  int wave = threadIdx.x >> 5, lane = threadIdx.x & 31;
  int token = blockIdx.x * 8 + wave;
  if (token >= L) return;
  const float* p = in + (size_t)token * CDIM;
  float v[8];
  float s = 0.f;
#pragma unroll
  for (int i = 0; i < 8; ++i) { v[i] = p[lane + i * 32]; s += v[i]; }
#pragma unroll
  for (int m = 16; m > 0; m >>= 1) s += __shfl_xor(s, m, 32);
  float mu = s * (1.0f / CDIM);
  float s2 = 0.f;
#pragma unroll
  for (int i = 0; i < 8; ++i) { float d = v[i] - mu; s2 += d * d; }
#pragma unroll
  for (int m = 16; m > 0; m >>= 1) s2 += __shfl_xor(s2, m, 32);
  float rstd = rsqrtf(s2 * (1.0f / CDIM) + 1e-5f);
  float* q = out + (size_t)token * CDIM;
#pragma unroll
  for (int i = 0; i < 8; ++i) {
    int c = lane + i * 32;
    q[c] = (v[i] - mu) * rstd * g[c] + b[c];
  }
}

// ----- tiled bf16 WMMA GEMM ------------------------------------------------
// C[M x N] = A[M x K](f32) @ Bw[K x N](bf16)  (+bias)(+residual)(+relu)
// Block: 64x64 tile, 128 threads = 4 waves in 2x2, BK = 64 (8 WMMA / stage).
// Epilogue variants are compile-time templates: branch-free straight-line
// code so residual loads batch under a single wait.

template <bool BIAS, bool RES, bool RELU>
__global__ __launch_bounds__(128) void gemm_bf16_kernel(
    const float* __restrict__ A, const unsigned short* __restrict__ Bw,
    float* __restrict__ C, const float* __restrict__ bias,
    const float* __restrict__ residual, int M, int N, int K) {
  __shared__ unsigned As[64 * 32];   // 64 rows x 64 bf16
  __shared__ unsigned Bs[64 * 32];   // transposed: 64 n-rows x 64 k bf16
  int tid  = threadIdx.x;
  int lane = tid & 31, wave = tid >> 5;
  int bm = blockIdx.y * 64, bn = blockIdx.x * 64;
  int wr = (wave >> 1) * 32, wc = (wave & 1) * 32;
  v8f acc[2][2] = {};

  for (int k0 = 0; k0 < K; k0 += 64) {
    // A: 64x64 f32 -> bf16.  1024 float4 / 128 threads = 8 x b128 loads.
#pragma unroll
    for (int i = 0; i < 8; ++i) {
      int u = tid * 8 + i;                 // float4 index 0..1023
      int r = u >> 4, q = u & 15;          // row, float4-within-row
      const float4 a4 = *(const float4*)(A + (size_t)(bm + r) * K + k0 + q * 4);
      As[r * 32 + q * 2]     = pack2(a4.x, a4.y);
      As[r * 32 + q * 2 + 1] = pack2(a4.z, a4.w);
    }
    // B: 64x64 bf16, transpose into Bs[n][k].  1024 uint2 / 128 thr = 8 each.
#pragma unroll
    for (int i = 0; i < 8; ++i) {
      int u = tid * 8 + i;                 // uint2 index 0..1023
      int kk = u >> 4, n4 = u & 15;        // k-row, 4-col group
      const uint2 w = *(const uint2*)(Bw + (size_t)(k0 + kk) * N + bn + n4 * 4);
      unsigned short* bsh = (unsigned short*)Bs;
      bsh[(size_t)(n4 * 4 + 0) * 64 + kk] = (unsigned short)(w.x & 0xFFFFu);
      bsh[(size_t)(n4 * 4 + 1) * 64 + kk] = (unsigned short)(w.x >> 16);
      bsh[(size_t)(n4 * 4 + 2) * 64 + kk] = (unsigned short)(w.y & 0xFFFFu);
      bsh[(size_t)(n4 * 4 + 3) * 64 + kk] = (unsigned short)(w.y >> 16);
    }
    __syncthreads();
#pragma unroll
    for (int ks = 0; ks < 2; ++ks) {
      v16bf af0 = ld_frag(As, wr,      32, ks * 16, lane);
      v16bf af1 = ld_frag(As, wr + 16, 32, ks * 16, lane);
      v16bf bf0 = ld_frag(Bs, wc,      32, ks * 16, lane);
      v16bf bf1 = ld_frag(Bs, wc + 16, 32, ks * 16, lane);
      acc[0][0] = wmma_bf(af0, bf0, acc[0][0]);
      acc[0][1] = wmma_bf(af0, bf1, acc[0][1]);
      acc[1][0] = wmma_bf(af1, bf0, acc[1][0]);
      acc[1][1] = wmma_bf(af1, bf1, acc[1][1]);
    }
    __syncthreads();
  }

  int h = lane >> 4, nn = lane & 15;
  float bias0 = 0.f, bias1 = 0.f;
  if (BIAS) {
    bias0 = bias[bn + wc + nn];
    bias1 = bias[bn + wc + 16 + nn];
  }
#pragma unroll
  for (int i = 0; i < 2; ++i)
#pragma unroll
    for (int j = 0; j < 2; ++j) {
      FragF d; d.v = acc[i][j];
      float bv = BIAS ? (j ? bias1 : bias0) : 0.f;
#pragma unroll
      for (int r = 0; r < 8; ++r) {
        int row = bm + wr + i * 16 + r + 8 * h;
        int col = bn + wc + j * 16 + nn;
        float val = d.f[r] + bv;
        if (RES)  val += residual[(size_t)row * N + col];
        if (RELU) val = fmaxf(val, 0.f);
        C[(size_t)row * N + col] = val;
      }
    }
}

// ----- fused windowed attention -------------------------------------------
// One block = one (image, window, head).  64 query tokens attend to 64 key
// tokens gathered from (kvH,kvW) with window stride `kstride` and circular
// wrap (self-attn: kstride=8, kv==q map; cross-attn: kstride=2, pooled map).
// The 1/sqrt(dh) scale is folded into the Q bf16 staging so the S-phase WMMA
// results feed straight into LDS stores (no WMMA->VALU hazard chain).

__device__ __forceinline__ int q_row(int img, int wy, int wx, int t, int qH, int qW) {
  int y = wy * 8 + (t >> 3);
  int x = wx * 8 + (t & 7);
  return (img * qH + y) * qW + x;
}
__device__ __forceinline__ int kv_row(int img, int wy, int wx, int t,
                                      int kstride, int kvH, int kvW) {
  int y = (wy * kstride + (t >> 3)) % kvH;
  int x = (wx * kstride + (t & 7)) % kvW;
  return (img * kvH + y) * kvW + x;
}

__global__ __launch_bounds__(128) void win_attn_kernel(
    const float* __restrict__ Qp, const float* __restrict__ Kp,
    const float* __restrict__ Vp, float* __restrict__ O,
    int qH, int qW, int kvH, int kvW, int kstride, float scale) {
  __shared__ unsigned Qs[64 * 16];   // 64 q   x 32 dh  bf16 (pre-scaled)
  __shared__ unsigned Ks[64 * 16];   // 64 key x 32 dh  bf16
  __shared__ unsigned Vs[32 * 32];   // 32 dh  x 64 key bf16 (transposed)
  __shared__ float    Ss[64 * 64];   // scores f32
  __shared__ unsigned Ps[64 * 32];   // softmax probs bf16, 64 q x 64 key

  int nWx = qW >> 3, nWy = qH >> 3;
  int winPerImg = nWx * nWy;
  int bid  = blockIdx.x;
  int head = bid & (HEADS - 1);
  int w    = (bid / HEADS) % winPerImg;
  int img  = bid / (HEADS * winPerImg);
  int wy = w / nWx, wx = w % nWx;
  int tid = threadIdx.x, lane = tid & 31, wave = tid >> 5;
  int co = head * DH;

  // Stage Q (pre-scaled) and K (64 x 32 each, row-major over dh), float2 loads.
#pragma unroll
  for (int i = 0; i < 8; ++i) {
    int u = tid * 8 + i;            // 0..1023
    int t = u >> 4, du = u & 15;    // token, dh dword
    const float2 qv = *(const float2*)(
        Qp + (size_t)q_row(img, wy, wx, t, qH, qW) * CDIM + co + du * 2);
    Qs[u] = pack2(qv.x * scale, qv.y * scale);
    const float2 kv = *(const float2*)(
        Kp + (size_t)kv_row(img, wy, wx, t, kstride, kvH, kvW) * CDIM + co + du * 2);
    Ks[u] = pack2(kv.x, kv.y);
  }
  // Stage V transposed: Vs[d][key] (gather, scalar loads)
#pragma unroll
  for (int i = 0; i < 8; ++i) {
    int u = tid * 8 + i;            // 0..1023
    int d = u >> 5, kp2 = u & 31;   // dh row, key dword
    int t0 = kp2 * 2, t1 = t0 + 1;
    float v0 = Vp[(size_t)kv_row(img, wy, wx, t0, kstride, kvH, kvW) * CDIM + co + d];
    float v1 = Vp[(size_t)kv_row(img, wy, wx, t1, kstride, kvH, kvW) * CDIM + co + d];
    Vs[u] = pack2(v0, v1);
  }
  __syncthreads();

  // S = (Q*scale) K^T : each wave owns a 16-row strip
  {
    v16bf aq = ld_frag(Qs, wave * 16, 16, 0, lane);
    int h = lane >> 4, nn = lane & 15;
#pragma unroll
    for (int j = 0; j < 4; ++j) {
      v8f s = {};
      v16bf bk = ld_frag(Ks, j * 16, 16, 0, lane);
      s = wmma_bf(aq, bk, s);
      FragF d; d.v = s;
#pragma unroll
      for (int r = 0; r < 8; ++r)
        Ss[(size_t)(wave * 16 + r + 8 * h) * 64 + j * 16 + nn] = d.f[r];
    }
  }
  __syncthreads();

  // Row softmax -> bf16 P
  if (tid < 64) {
    float* row = Ss + (size_t)tid * 64;
    float mx = -3.402823466e+38f;
#pragma unroll
    for (int c = 0; c < 64; ++c) mx = fmaxf(mx, row[c]);
    float sum = 0.f;
#pragma unroll
    for (int c = 0; c < 64; ++c) { float e = __expf(row[c] - mx); row[c] = e; sum += e; }
    float inv = 1.0f / sum;
#pragma unroll
    for (int cu = 0; cu < 32; ++cu)
      Ps[tid * 32 + cu] = pack2(row[cu * 2] * inv, row[cu * 2 + 1] * inv);
  }
  __syncthreads();

  // O = P V : 16-row strip per wave, dh = 32 (2 n-tiles), K = 64 (2 k-steps)
  {
    v8f oacc[2] = {};
#pragma unroll
    for (int ks = 0; ks < 2; ++ks) {
      v16bf ap = ld_frag(Ps, wave * 16, 32, ks * 16, lane);
#pragma unroll
      for (int j = 0; j < 2; ++j) {
        v16bf bv = ld_frag(Vs, j * 16, 32, ks * 16, lane);
        oacc[j] = wmma_bf(ap, bv, oacc[j]);
      }
    }
    int h = lane >> 4, nn = lane & 15;
#pragma unroll
    for (int j = 0; j < 2; ++j) {
      FragF d; d.v = oacc[j];
#pragma unroll
      for (int r = 0; r < 8; ++r) {
        int t = wave * 16 + r + 8 * h;
        O[(size_t)q_row(img, wy, wx, t, qH, qW) * CDIM + co + j * 16 + nn] = d.f[r];
      }
    }
  }
}

// ---------------------------------------------------------------------------
// Host orchestration
// ---------------------------------------------------------------------------

static inline void launch_gemm(const float* A, const unsigned short* Bw, float* C,
                               const float* bias, const float* residual,
                               int M, int N, int K, int relu, hipStream_t s) {
  dim3 grid(N / 64, M / 64), block(128);
  if (!bias && !residual && !relu)
    gemm_bf16_kernel<false, false, false><<<grid, block, 0, s>>>(A, Bw, C, bias, residual, M, N, K);
  else if (bias && residual && !relu)
    gemm_bf16_kernel<true, true, false><<<grid, block, 0, s>>>(A, Bw, C, bias, residual, M, N, K);
  else if (bias && !residual && relu)
    gemm_bf16_kernel<true, false, true><<<grid, block, 0, s>>>(A, Bw, C, bias, residual, M, N, K);
  else
    gemm_bf16_kernel<true, true, true><<<grid, block, 0, s>>>(A, Bw, C, bias, residual, M, N, K);
}

extern "C" void kernel_launch(void* const* d_in, const int* in_sizes, int n_in,
                              void* d_out, int out_size, void* d_ws, size_t ws_size,
                              hipStream_t stream) {
  (void)in_sizes; (void)n_in; (void)out_size; (void)ws_size;

  const float* x        = (const float*)d_in[0];
  const float* sa_wq    = (const float*)d_in[1];
  const float* sa_wk    = (const float*)d_in[2];
  const float* sa_wv    = (const float*)d_in[3];
  const float* sa_wf    = (const float*)d_in[4];
  const float* sa_bf    = (const float*)d_in[5];
  const float* ca_wq    = (const float*)d_in[6];
  const float* ca_wk    = (const float*)d_in[7];
  const float* ca_wv    = (const float*)d_in[8];
  const float* ca_wf    = (const float*)d_in[9];
  const float* ca_bf    = (const float*)d_in[10];
  const float* ln_self_g = (const float*)d_in[11];
  const float* ln_self_b = (const float*)d_in[12];
  const float* ln_cross_g = (const float*)d_in[13];
  const float* ln_cross_b = (const float*)d_in[14];
  const float* ffn_w1   = (const float*)d_in[15];
  const float* ffn_b1   = (const float*)d_in[16];
  const float* ffn_w2   = (const float*)d_in[17];
  const float* ffn_b2   = (const float*)d_in[18];
  const float* ln_out_g = (const float*)d_in[19];
  const float* ln_out_b = (const float*)d_in[20];

  const int Nimg = 8, Himg = 128, Wimg = 128, H1 = 32, W1 = 32;
  const size_t L0 = (size_t)Nimg * Himg * Wimg;   // 131072
  const size_t L1 = (size_t)Nimg * H1 * W1;       // 8192
  const size_t SZ0 = L0 * CDIM * sizeof(float);   // 128 MiB
  const size_t SZ1 = L1 * CDIM * sizeof(float);

  char* ws = (char*)d_ws;
  float* A0  = (float*)(ws + 0 * SZ0);
  float* B0  = (float*)(ws + 1 * SZ0);
  float* B1  = (float*)(ws + 2 * SZ0);
  float* B2  = (float*)(ws + 3 * SZ0);
  float* B3  = (float*)(ws + 4 * SZ0);
  float* B4  = (float*)(ws + 5 * SZ0);
  float* XA0 = (float*)(ws + 6 * SZ0);
  float* QF  = (float*)(ws + 7 * SZ0);
  float* Hbuf = B0;                    // L0 x 1024: aliases dead B0..B3 region
  char* sm = ws + 8 * SZ0;
  float* A1  = (float*)(sm + 0 * SZ1);
  float* q1  = (float*)(sm + 1 * SZ1);
  float* k1  = (float*)(sm + 2 * SZ1);
  float* v1  = (float*)(sm + 3 * SZ1);
  float* o1  = (float*)(sm + 4 * SZ1);
  float* t1  = (float*)(sm + 5 * SZ1);
  float* XA1 = (float*)(sm + 6 * SZ1);
  unsigned short* wb = (unsigned short*)(sm + 7 * SZ1);
  unsigned short* w_saq = wb;              // each 256*256
  unsigned short* w_sak = wb + 65536;
  unsigned short* w_sav = wb + 2 * 65536;
  unsigned short* w_saf = wb + 3 * 65536;
  unsigned short* w_caq = wb + 4 * 65536;
  unsigned short* w_cak = wb + 5 * 65536;
  unsigned short* w_cav = wb + 6 * 65536;
  unsigned short* w_caf = wb + 7 * 65536;
  unsigned short* w_f1  = wb + 8 * 65536;              // 256x1024
  unsigned short* w_f2  = wb + 8 * 65536 + 262144;     // 1024x256

  const float scale = 0.17677669529663687f;            // 1/sqrt(32)
  const size_t totalElems = L0 * CDIM;                 // 33.5M

  // 0) input NCHW -> NHWC
  nchw_to_nhwc_kernel<<<dim3(totalElems / 256), dim3(256), 0, stream>>>(
      x, A0, Nimg, CDIM, Himg, Wimg);

  // 1) weights -> bf16
  f32_to_bf16_kernel<<<dim3(256), dim3(256), 0, stream>>>(sa_wq, w_saq, 65536);
  f32_to_bf16_kernel<<<dim3(256), dim3(256), 0, stream>>>(sa_wk, w_sak, 65536);
  f32_to_bf16_kernel<<<dim3(256), dim3(256), 0, stream>>>(sa_wv, w_sav, 65536);
  f32_to_bf16_kernel<<<dim3(256), dim3(256), 0, stream>>>(sa_wf, w_saf, 65536);
  f32_to_bf16_kernel<<<dim3(256), dim3(256), 0, stream>>>(ca_wq, w_caq, 65536);
  f32_to_bf16_kernel<<<dim3(256), dim3(256), 0, stream>>>(ca_wk, w_cak, 65536);
  f32_to_bf16_kernel<<<dim3(256), dim3(256), 0, stream>>>(ca_wv, w_cav, 65536);
  f32_to_bf16_kernel<<<dim3(256), dim3(256), 0, stream>>>(ca_wf, w_caf, 65536);
  f32_to_bf16_kernel<<<dim3(1024), dim3(256), 0, stream>>>(ffn_w1, w_f1, 262144);
  f32_to_bf16_kernel<<<dim3(1024), dim3(256), 0, stream>>>(ffn_w2, w_f2, 262144);

  // 2) level-0 self-attention
  launch_gemm(A0, w_saq, B0, nullptr, nullptr, (int)L0, CDIM, CDIM, 0, stream);
  launch_gemm(A0, w_sak, B1, nullptr, nullptr, (int)L0, CDIM, CDIM, 0, stream);
  launch_gemm(A0, w_sav, B2, nullptr, nullptr, (int)L0, CDIM, CDIM, 0, stream);
  win_attn_kernel<<<dim3(Nimg * 256 * HEADS), dim3(128), 0, stream>>>(
      B0, B1, B2, B3, Himg, Wimg, Himg, Wimg, 8, scale);
  launch_gemm(B3, w_saf, B4, sa_bf, A0, (int)L0, CDIM, CDIM, 0, stream);
  layernorm_kernel<<<dim3(L0 / 8), dim3(256), 0, stream>>>(
      B4, XA0, ln_self_g, ln_self_b, (int)L0);

  // 3) pool to level 1
  avgpool4_kernel<<<dim3(L1 * CDIM / 256), dim3(256), 0, stream>>>(
      XA0, A1, Nimg, Himg, Wimg);

  // 4) level-1 self-attention
  launch_gemm(A1, w_saq, q1, nullptr, nullptr, (int)L1, CDIM, CDIM, 0, stream);
  launch_gemm(A1, w_sak, k1, nullptr, nullptr, (int)L1, CDIM, CDIM, 0, stream);
  launch_gemm(A1, w_sav, v1, nullptr, nullptr, (int)L1, CDIM, CDIM, 0, stream);
  win_attn_kernel<<<dim3(Nimg * 16 * HEADS), dim3(128), 0, stream>>>(
      q1, k1, v1, o1, H1, W1, H1, W1, 8, scale);
  launch_gemm(o1, w_saf, t1, sa_bf, A1, (int)L1, CDIM, CDIM, 0, stream);
  layernorm_kernel<<<dim3(L1 / 8), dim3(256), 0, stream>>>(
      t1, XA1, ln_self_g, ln_self_b, (int)L1);

  // 5) cross-attention (queries: level-0 windows; keys/values: strided
  //    overlapping 8x8 patches of the pooled 32x32 map, circular wrap)
  launch_gemm(XA0, w_caq, B0, nullptr, nullptr, (int)L0, CDIM, CDIM, 0, stream);
  launch_gemm(XA1, w_cak, k1, nullptr, nullptr, (int)L1, CDIM, CDIM, 0, stream);
  launch_gemm(XA1, w_cav, v1, nullptr, nullptr, (int)L1, CDIM, CDIM, 0, stream);
  win_attn_kernel<<<dim3(Nimg * 256 * HEADS), dim3(128), 0, stream>>>(
      B0, k1, v1, B3, Himg, Wimg, H1, W1, 2, scale);
  launch_gemm(B3, w_caf, B4, ca_bf, XA0, (int)L0, CDIM, CDIM, 0, stream);
  layernorm_kernel<<<dim3(L0 / 8), dim3(256), 0, stream>>>(
      B4, QF, ln_cross_g, ln_cross_b, (int)L0);

  // 6) FFN: relu(QF @ W1 + b1) @ W2 + b2 + QF, then LN(out), then NHWC->NCHW
  launch_gemm(QF, w_f1, Hbuf, ffn_b1, nullptr, (int)L0, 1024, CDIM, 1, stream);
  launch_gemm(Hbuf, w_f2, B4, ffn_b2, QF, (int)L0, CDIM, 1024, 0, stream);
  layernorm_kernel<<<dim3(L0 / 8), dim3(256), 0, stream>>>(
      B4, QF, ln_out_g, ln_out_b, (int)L0);
  nhwc_to_nchw_kernel<<<dim3(totalElems / 256), dim3(256), 0, stream>>>(
      QF, (float*)d_out, Nimg, CDIM, Himg, Wimg);
}
